// SubMarginLinear_74363063763040
// MI455X (gfx1250) — compile-verified, matrix operand order checked by reading, not run on hardware
//
#include <hip/hip_runtime.h>
#include <hip/hip_bf16.h>

// ---------------------------------------------------------------------------
//   x : [B=512, D=512] f32 ; W : [D=512, N=17982] f32 (N = C*K)
//   out[b,n] = (x[b,:]/||x[b,:]||) . (W[:,n]/||W[:,n]||)
// fp32 norms -> bf16 v_wmma_f32_16x16x32_bf16 GEMM -> epilogue * 1/||W[:,n]||
// A = pre-normalized x (bf16), B = W transposed to [N][D] bf16 so both GEMM
// stage paths are aligned 16B copies (async global->LDS, ASYNCcnt-tracked).
// ---------------------------------------------------------------------------

typedef __bf16 bf16_t;
typedef __attribute__((ext_vector_type(16))) __bf16 v16bf;
typedef __attribute__((ext_vector_type(8)))  float  v8f;
typedef int v4i_gcc __attribute__((vector_size(4 * sizeof(int))));

#define DDIM   512
#define BDIM   512
#define NOUT   17982
#define NPAD   18176        // 71 * 256
#define NBLK   71           // N blocks of 256
#define LDT    40           // LDS k-stride (16B aligned, bank-conflict breaking)
#define EPS    1e-8f

// ---- CDNA5 async global->LDS path (ASYNCcnt), guarded -----------------------
#if defined(__has_builtin)
#if __has_builtin(__builtin_amdgcn_global_load_async_to_lds_b128) && \
    __has_builtin(__builtin_amdgcn_s_wait_asynccnt)
#define HAVE_ASYNC 1
#endif
#endif
#ifndef HAVE_ASYNC
#define HAVE_ASYNC 0
#endif

__device__ __forceinline__ void cp16(const bf16_t* g, bf16_t* l)
{
#if HAVE_ASYNC
    // Signature (from toolchain diagnostic): (v4i addrspace(1)*, v4i addrspace(3)*, imm, imm)
    __builtin_amdgcn_global_load_async_to_lds_b128(
        (__attribute__((address_space(1))) v4i_gcc*)(unsigned long long)g,
        (__attribute__((address_space(3))) v4i_gcc*)(unsigned)(unsigned long long)l,
        0, 0);
#else
    *(uint4*)l = *(const uint4*)g;
#endif
}

// ---------------------------------------------------------------------------
// Kernel 1: row-normalize x into bf16
// ---------------------------------------------------------------------------
__global__ __launch_bounds__(256) void normalize_x_kernel(
    const float* __restrict__ x, bf16_t* __restrict__ xb)
{
    __shared__ float red[256];
    const int t = threadIdx.x;
    const int b = blockIdx.x;

    const float a0 = x[b * DDIM + t];
    const float a1 = x[b * DDIM + t + 256];
    red[t] = a0 * a0 + a1 * a1;
    __syncthreads();
    #pragma unroll
    for (int s = 128; s > 0; s >>= 1) {
        if (t < s) red[t] += red[t + s];
        __syncthreads();
    }
    const float scale = 1.0f / fmaxf(sqrtf(red[0]), EPS);
    xb[b * DDIM + t]       = (bf16_t)(a0 * scale);
    xb[b * DDIM + t + 256] = (bf16_t)(a1 * scale);
}

// ---------------------------------------------------------------------------
// Kernel 2: per-column 1/||W[:,n]|| + bf16 TRANSPOSED copy Wt[n][d].
// One block = 128 columns. Norm pass is coalesced over n; transpose goes
// through an LDS tile (second read of the 256KB slice hits L2).
// ---------------------------------------------------------------------------
__global__ __launch_bounds__(256) void prep_w_kernel(
    const float* __restrict__ W, bf16_t* __restrict__ Wt, float* __restrict__ RW)
{
    __shared__ float  red[256];
    __shared__ bf16_t tile[64 * 136];     // [d_local][n], padded row 136
    const int tid = threadIdx.x;
    const int n0  = blockIdx.x * 128;
    const int c   = n0 + (tid & 127);
    const int dh  = tid >> 7;

    float ss = 0.0f;
    for (int d = dh * 256; d < dh * 256 + 256; ++d) {
        const float w = (c < NOUT) ? W[d * NOUT + c] : 0.0f;
        ss += w * w;
    }
    red[tid] = ss;
    __syncthreads();
    if (tid < 128) {
        const float tot = red[tid] + red[tid + 128];
        RW[c] = (c < NOUT) ? (1.0f / fmaxf(sqrtf(tot), EPS)) : 0.0f;
    }

    // transpose 64-d slabs: coalesced read [d][n] -> LDS -> 16B writes [n][d]
    for (int dc = 0; dc < DDIM; dc += 64) {
        __syncthreads();
        #pragma unroll
        for (int j = 0; j < 32; ++j) {
            const int e  = tid + j * 256;
            const int dl = e >> 7, n = e & 127;
            const float w = ((n0 + n) < NOUT) ? W[(dc + dl) * NOUT + n0 + n] : 0.0f;
            tile[dl * 136 + n] = (bf16_t)w;
        }
        __syncthreads();
        #pragma unroll
        for (int j = 0; j < 4; ++j) {
            const int u  = tid + j * 256;
            const int nl = u >> 3, d8 = (u & 7) * 8;
            union { uint4 q; bf16_t h[8]; } pk;
            #pragma unroll
            for (int e2 = 0; e2 < 8; ++e2) pk.h[e2] = tile[(d8 + e2) * 136 + nl];
            *(uint4*)(Wt + (n0 + nl) * DDIM + dc + d8) = pk.q;
        }
    }
}

// ---------------------------------------------------------------------------
// WMMA fragment load (ISA 7.12.2 16-bit layout): lanes 0-15 hold K[0..7] in
// elems 0-7 and K[16..23] in elems 8-15; lanes 16-31 hold K[8..15], K[24..31].
// ---------------------------------------------------------------------------
union FragCast { uint4 u4[2]; v16bf v; };

__device__ __forceinline__ v16bf ld_frag(const bf16_t* p, int half)
{
    FragCast f;
    f.u4[0] = *(const uint4*)(p + (half << 3));
    f.u4[1] = *(const uint4*)(p + 16 + (half << 3));
    return f.v;
}

// ---------------------------------------------------------------------------
// Kernel 3: bf16 WMMA GEMM. Block 128x256, 8 waves (2x4), 64x64 per wave,
// K-step 32, double-buffered LDS fed by async global->LDS b128 copies.
// ---------------------------------------------------------------------------
__global__ __launch_bounds__(256) void cosine_gemm_kernel(
    const bf16_t* __restrict__ xb,   // [BDIM][DDIM] pre-normalized
    const bf16_t* __restrict__ Wt,   // [NPAD][DDIM] raw magnitudes
    const float*  __restrict__ RW,   // [NPAD] 1/||W[:,n]||
    float*        __restrict__ out)  // [BDIM][NOUT]
{
    __shared__ bf16_t As[2][128 * LDT];   // [row][k]
    __shared__ bf16_t Bs[2][256 * LDT];   // [col][k]

    const int tid  = threadIdx.x;
    const int n0   = blockIdx.x * 256;
    const int m0   = blockIdx.y * 128;
    const int wave = tid >> 5;
    const int lane = tid & 31;
    const int half = lane >> 4;
    const int r    = lane & 15;
    const int wm   = (wave >> 2) * 64;
    const int wn   = (wave & 3)  * 64;
    const int srow = tid >> 2;            // staging: 64 rows / pass
    const int scol = (tid & 3) * 8;       // 16B chunk within k

    v8f acc[4][4] = {};

    auto stage = [&](int kt, int buf) {   // 6 x b128 per thread (2 A + 4 B)
        const int k0 = kt * 32;
        #pragma unroll
        for (int p = 0; p < 2; ++p) {
            const int row = srow + p * 64;
            cp16(xb + (m0 + row) * DDIM + k0 + scol, &As[buf][row * LDT + scol]);
        }
        #pragma unroll
        for (int p = 0; p < 4; ++p) {
            const int row = srow + p * 64;
            cp16(Wt + (n0 + row) * DDIM + k0 + scol, &Bs[buf][row * LDT + scol]);
        }
    };

    stage(0, 0);
    for (int kt = 0; kt < 16; ++kt) {
        const int cur = kt & 1;
        if (kt < 15) stage(kt + 1, cur ^ 1);     // prefetch next tile into other buffer
#if HAVE_ASYNC
        if (kt < 15) __builtin_amdgcn_s_wait_asynccnt(6);  // only next tile in flight
        else         __builtin_amdgcn_s_wait_asynccnt(0);
#endif
        __syncthreads();

        v16bf af[4], bfr[4];
        #pragma unroll
        for (int mt = 0; mt < 4; ++mt)
            af[mt] = ld_frag(&As[cur][(wm + mt * 16 + r) * LDT], half);
        #pragma unroll
        for (int nt = 0; nt < 4; ++nt)
            bfr[nt] = ld_frag(&Bs[cur][(wn + nt * 16 + r) * LDT], half);

        #pragma unroll
        for (int mt = 0; mt < 4; ++mt)
            #pragma unroll
            for (int nt = 0; nt < 4; ++nt)
                acc[mt][nt] = __builtin_amdgcn_wmma_f32_16x16x32_bf16(
                    false, af[mt], false, bfr[nt],
                    (short)0, acc[mt][nt], false, false);
        __syncthreads();                         // reads done before buffer reuse
    }

    #pragma unroll
    for (int nt = 0; nt < 4; ++nt) {
        const int gn = n0 + wn + nt * 16 + r;
        const bool ok = gn < NOUT;
        const float rwv = ok ? RW[gn] : 0.0f;
        #pragma unroll
        for (int mt = 0; mt < 4; ++mt) {
            const int gm = m0 + wm + mt * 16 + half * 8;
            #pragma unroll
            for (int v = 0; v < 8; ++v) {
                if (ok) out[(gm + v) * NOUT + gn] = acc[mt][nt][v] * rwv;
            }
        }
    }
}

// ---------------------------------------------------------------------------
extern "C" void kernel_launch(void* const* d_in, const int* in_sizes, int n_in,
                              void* d_out, int out_size, void* d_ws, size_t ws_size,
                              hipStream_t stream)
{
    const float* x = (const float*)d_in[0];
    const float* W = (const float*)d_in[1];
    float* out = (float*)d_out;

    char* ws = (char*)d_ws;
    const size_t XB_BYTES = (size_t)BDIM * DDIM * sizeof(bf16_t);   // 512 KB
    const size_t WT_BYTES = (size_t)NPAD * DDIM * sizeof(bf16_t);   // ~18.6 MB
    bf16_t* xb = (bf16_t*)ws;
    bf16_t* Wt = (bf16_t*)(ws + XB_BYTES);
    float*  RW = (float*)(ws + XB_BYTES + WT_BYTES);

    normalize_x_kernel<<<BDIM, 256, 0, stream>>>(x, xb);
    prep_w_kernel<<<NPAD / 128, 256, 0, stream>>>(W, Wt, RW);
    cosine_gemm_kernel<<<dim3(NBLK, BDIM / 128), 256, 0, stream>>>(xb, Wt, RW, out);
}
